// QCGN2oEI_Minimal_27298812133433
// MI455X (gfx1250) — matrix operand order repaired and checked by reading, not compiled, not run on hardware
//
#include <hip/hip_runtime.h>
#include <hip/hip_bf16.h>

// ---------------- types for WMMA ----------------
typedef __bf16 v16bf __attribute__((ext_vector_type(16)));
typedef float  v8f   __attribute__((ext_vector_type(8)));
typedef int    gat_v4i __attribute__((ext_vector_type(4)));

union GatFrag { v16bf v; unsigned short s[16]; };

// ---------------- async global->LDS (gfx1250) ----------------
#ifndef GAT_HAS_ASYNC
#  if defined(__has_builtin)
#    if __has_builtin(__builtin_amdgcn_global_load_async_to_lds_b128) && \
        __has_builtin(__builtin_amdgcn_s_wait_asynccnt)
#      define GAT_HAS_ASYNC 1
#    endif
#  endif
#endif
#ifndef GAT_HAS_ASYNC
#  define GAT_HAS_ASYNC 0
#endif

#if GAT_HAS_ASYNC
__device__ __forceinline__ void gat_async_b128(const void* g, void* l) {
    // signature (from hipcc diagnostic): param is generic `int4*` (ext-vector)
    __builtin_amdgcn_global_load_async_to_lds_b128(
        (gat_v4i*)g, (gat_v4i*)l, 0, 0);
}
#endif

// ---------------- small device helpers ----------------
__device__ __forceinline__ unsigned short gat_f2bf(float f) {
    unsigned int u = __float_as_uint(f);
    unsigned int r = (u + 0x7FFFu + ((u >> 16) & 1u)) >> 16;   // RNE
    return (unsigned short)r;
}
__device__ __forceinline__ float gat_bf2f(unsigned short s) {
    return __uint_as_float(((unsigned int)s) << 16);
}
__device__ __forceinline__ float gat_elu(float x) {
    return x > 0.0f ? x : (expf(x) - 1.0f);
}
// order-preserving float<->uint map for atomic max
__device__ __forceinline__ unsigned int gat_ordf(float f) {
    unsigned int u = __float_as_uint(f);
    return (u & 0x80000000u) ? ~u : (u | 0x80000000u);
}
__device__ __forceinline__ float gat_unordf(unsigned int u) {
    return (u & 0x80000000u) ? __uint_as_float(u & 0x7FFFFFFFu)
                             : __uint_as_float(~u);
}

#define GAT_D   768
#define GAT_H   24
#define GAT_C   32
#define GAT_L   14

// ================= utility kernels =================
__global__ void gat_zero_f32(float* __restrict__ p, size_t n) {
    size_t i = (size_t)blockIdx.x * blockDim.x + threadIdx.x;
    if (i < n) p[i] = 0.0f;
}

// f32 [.., K, N] -> bf16 transposed [.., N, K] (weights become N-major so the
// GEMM B tile stages into LDS with pure b128 copies, no scalar transpose).
__global__ void gat_cvt_bf16_T(const float* __restrict__ src,
                               unsigned short* __restrict__ dst, size_t total) {
    size_t t = (size_t)blockIdx.x * blockDim.x + threadIdx.x;
    if (t >= total) return;
    size_t mat = t / ((size_t)GAT_D * GAT_D);
    size_t rem = t % ((size_t)GAT_D * GAT_D);
    int k = (int)(rem / GAT_D);
    int n = (int)(rem % GAT_D);
    dst[mat * GAT_D * GAT_D + (size_t)n * GAT_D + k] = gat_f2bf(src[t]);
}

// ELU(x @ W + b) for small K (encoders). Writes f32 (optional) and bf16.
__global__ void gat_encoder(const float* __restrict__ X,
                            const float* __restrict__ W,
                            const float* __restrict__ b,
                            float* __restrict__ outF,
                            unsigned short* __restrict__ outBf,
                            int M, int Kd, int Nd) {
    size_t t = (size_t)blockIdx.x * blockDim.x + threadIdx.x;
    if (t >= (size_t)M * Nd) return;
    int i = (int)(t / Nd);
    int j = (int)(t % Nd);
    float acc = b[j];
    const float* xr = X + (size_t)i * Kd;
    for (int k = 0; k < Kd; ++k) acc += xr[k] * W[(size_t)k * Nd + j];
    acc = gat_elu(acc);
    if (outF) outF[t] = acc;
    outBf[t] = gat_f2bf(acc);
}

// ================= WMMA GEMM =================
// C = A(bf16, MxK row-major) @ B (given TRANSPOSED: BT is NnxK row-major) + bias
// 128 threads = 4 waves. Block tile 128(M) x 64(N); each wave owns 32x64
// (2 M-subtiles x 4 N-subtiles = 8 accumulators). K stepped by 32.
// Requires: K % 32 == 0, Nn % 64 == 0 (true for all uses here: K=768, Nn=768).
#define GT_BM 128
#define GT_BN 64
#define GT_K  32

__global__ __launch_bounds__(128)
void gat_wmma_gemm(const unsigned short* __restrict__ A,
                   const unsigned short* __restrict__ BT,
                   const float* __restrict__ bias,
                   float* __restrict__ Cf,
                   unsigned short* __restrict__ Cbf,
                   int M, int K, int Nn) {
    __shared__ __align__(16) unsigned short sA[GT_BM][GT_K];   // [m][k]
    __shared__ __align__(16) unsigned short sB[GT_BN][GT_K];   // [n][k]

    const int tid  = threadIdx.x;
    const int wave = tid >> 5;
    const int lane = tid & 31;
    const int bx   = blockIdx.x;   // N tile
    const int by   = blockIdx.y;   // M tile

    const bool fullM = (by * GT_BM + GT_BM) <= M;   // uniform over the block

    v8f acc[8] = {};

    for (int k0 = 0; k0 < K; k0 += GT_K) {
        // ---- stage A tile 128x32: one row (64B) per thread ----
        // ---- stage B tile 64x32: half row (32B) per thread ----
        const int ar  = tid;                       // 0..127
        const int arow = by * GT_BM + ar;
        const int bn  = tid >> 1;                  // 0..63
        const int bc0 = (tid & 1) << 4;            // 0 or 16
        const int bgn = bx * GT_BN + bn;
        const unsigned short* gA = A  + (size_t)arow * K + k0;
        const unsigned short* gB = BT + (size_t)bgn  * K + k0 + bc0;

        if (fullM) {
#if GAT_HAS_ASYNC
            #pragma unroll
            for (int c = 0; c < 4; ++c)
                gat_async_b128(gA + c * 8, &sA[ar][c * 8]);
            #pragma unroll
            for (int c = 0; c < 2; ++c)
                gat_async_b128(gB + c * 8, &sB[bn][bc0 + c * 8]);
            __builtin_amdgcn_s_wait_asynccnt(0);
#else
            {
                const uint4* g = (const uint4*)gA;
                uint4* d = (uint4*)(&sA[ar][0]);
                d[0] = g[0]; d[1] = g[1]; d[2] = g[2]; d[3] = g[3];
            }
            {
                const uint4* g = (const uint4*)gB;
                uint4* d = (uint4*)(&sB[bn][bc0]);
                d[0] = g[0]; d[1] = g[1];
            }
#endif
        } else {
            uint4 z = make_uint4(0u, 0u, 0u, 0u);
            uint4 q0 = z, q1 = z, q2 = z, q3 = z;
            if (arow < M) {
                const uint4* g = (const uint4*)gA;
                q0 = g[0]; q1 = g[1]; q2 = g[2]; q3 = g[3];
            }
            uint4* d = (uint4*)(&sA[ar][0]);
            d[0] = q0; d[1] = q1; d[2] = q2; d[3] = q3;
            const uint4* g = (const uint4*)gB;
            uint4* db = (uint4*)(&sB[bn][bc0]);
            db[0] = g[0]; db[1] = g[1];
        }
        __syncthreads();

        // ---- A fragments (ISA 16-bit A 16x32 layout), 2 M-subtiles ----
        GatFrag af[2];
        {
            int ko = (lane < 16) ? 0 : 8;
            #pragma unroll
            for (int mh = 0; mh < 2; ++mh) {
                int r = wave * 32 + mh * 16 + (lane & 15);
                #pragma unroll
                for (int j = 0; j < 8; ++j) {
                    af[mh].s[j]     = sA[r][ko + j];
                    af[mh].s[8 + j] = sA[r][16 + ko + j];
                }
            }
        }
        // ---- 4 N-subtiles x 2 M-subtiles = 8 WMMAs per K step ----
        #pragma unroll
        for (int nt = 0; nt < 4; ++nt) {
            GatFrag bf;
            int nn = nt * 16 + (lane & 15);
            int kb = (lane < 16) ? 0 : 8;
            #pragma unroll
            for (int j = 0; j < 8; ++j) {
                bf.s[j]     = sB[nn][kb + j];
                bf.s[8 + j] = sB[nn][16 + kb + j];
            }
            #pragma unroll
            for (int mh = 0; mh < 2; ++mh) {
                acc[mh * 4 + nt] = __builtin_amdgcn_wmma_f32_16x16x32_bf16(
                    false, af[mh].v, false, bf.v, (short)0,
                    acc[mh * 4 + nt], false, false);
            }
        }
        __syncthreads();
    }

    // ---- store (ISA 32-bit C/D 16x16 layout) ----
    #pragma unroll
    for (int nt = 0; nt < 4; ++nt) {
        int nIdx = bx * GT_BN + nt * 16 + (lane & 15);
        float bv = bias ? bias[nIdx] : 0.0f;
        #pragma unroll
        for (int mh = 0; mh < 2; ++mh) {
            #pragma unroll
            for (int j = 0; j < 8; ++j) {
                int mIdx = by * GT_BM + wave * 32 + mh * 16 +
                           ((lane < 16) ? j : 8 + j);
                if (mIdx < M) {
                    float v = acc[mh * 4 + nt][j] + bv;
                    size_t o = (size_t)mIdx * Nn + nIdx;
                    if (Cf)  Cf[o]  = v;
                    if (Cbf) Cbf[o] = gat_f2bf(v);
                }
            }
        }
    }
}

// ================= edge-side kernels =================
// One block (256 thr = 8 waves) per edge; wave w covers heads w, w+8, w+16;
// lane = channel c (wave32 == C == 32).
__global__ __launch_bounds__(256)
void gat_edge_logits(const int* __restrict__ ei,
                     const float* __restrict__ xl,
                     const float* __restrict__ xr,
                     const unsigned short* __restrict__ ee,
                     const float* __restrict__ att,     // layer slice [H*C]
                     float* __restrict__ logits,        // [E*H]
                     unsigned int* __restrict__ lmax,   // [N*H] ordered-uint
                     int E) {
    int e = blockIdx.x;
    if (e >= E) return;
    int lane = threadIdx.x & 31;
    int wv   = threadIdx.x >> 5;
    int src = ei[e];
    int dst = ei[E + e];
    const float* xls = xl + (size_t)src * GAT_D;
    const float* xrd = xr + (size_t)dst * GAT_D;
    const unsigned short* ees = ee + (size_t)e * GAT_D;
    for (int h = wv; h < GAT_H; h += 8) {
        int idx = h * GAT_C + lane;
        float m = xls[idx] + xrd[idx] + gat_bf2f(ees[idx]);
        m = (m > 0.0f) ? m : 0.2f * m;            // leaky_relu(0.2)
        float v = m * att[idx];
        #pragma unroll
        for (int o = 16; o > 0; o >>= 1) v += __shfl_xor(v, o, 32);
        if (lane == 0) {
            logits[(size_t)e * GAT_H + h] = v;
            atomicMax(&lmax[(size_t)dst * GAT_H + h], gat_ordf(v));
        }
    }
}

// p = exp(logit - max); accumulate denominator. One thread per (e,h).
__global__ void gat_edge_softmax(const int* __restrict__ ei,
                                 float* __restrict__ logits,       // in: logit, out: p
                                 const unsigned int* __restrict__ lmax,
                                 float* __restrict__ denom,
                                 int E) {
    size_t t = (size_t)blockIdx.x * blockDim.x + threadIdx.x;
    if (t >= (size_t)E * GAT_H) return;
    int e = (int)(t / GAT_H);
    int h = (int)(t % GAT_H);
    int dst = ei[E + e];
    float lm = gat_unordf(lmax[(size_t)dst * GAT_H + h]);
    float p  = expf(logits[t] - lm);
    logits[t] = p;
    atomicAdd(&denom[(size_t)dst * GAT_H + h], p);
}

// out[dst] += alpha * xl[src] ; block per edge.
__global__ __launch_bounds__(256)
void gat_edge_aggregate(const int* __restrict__ ei,
                        const float* __restrict__ p,       // [E*H]
                        const float* __restrict__ denom,   // [N*H]
                        const float* __restrict__ xl,
                        float* __restrict__ agg,           // [N*D]
                        int E) {
    int e = blockIdx.x;
    if (e >= E) return;
    int lane = threadIdx.x & 31;
    int wv   = threadIdx.x >> 5;
    int src = ei[e];
    int dst = ei[E + e];
    const float* xls = xl + (size_t)src * GAT_D;
    float* aggd = agg + (size_t)dst * GAT_D;
    for (int h = wv; h < GAT_H; h += 8) {
        float a = p[(size_t)e * GAT_H + h] /
                  (denom[(size_t)dst * GAT_H + h] + 1e-16f);
        int idx = h * GAT_C + lane;
        atomicAdd(&aggd[idx], a * xls[idx]);
    }
}

// h = ELU(agg + bg + hres); writes f32 + bf16
__global__ void gat_node_update(const float* __restrict__ agg,
                                const float* __restrict__ bg,     // [D]
                                const float* __restrict__ hres,
                                float* __restrict__ hF,
                                unsigned short* __restrict__ hBf,
                                size_t total) {
    size_t t = (size_t)blockIdx.x * blockDim.x + threadIdx.x;
    if (t >= total) return;
    int j = (int)(t % GAT_D);
    float v = gat_elu(agg[t] + bg[j] + hres[t]);
    hF[t]  = v;
    hBf[t] = gat_f2bf(v);
}

// ================= head kernels =================
__global__ __launch_bounds__(256)
void gat_col_mean(const float* __restrict__ h, float* __restrict__ gr, int n) {
    __shared__ float red[256];
    int j = blockIdx.x;
    float s = 0.0f;
    for (int r = threadIdx.x; r < n; r += 256) s += h[(size_t)r * GAT_D + j];
    red[threadIdx.x] = s; __syncthreads();
    for (int k = 128; k > 0; k >>= 1) {
        if (threadIdx.x < k) red[threadIdx.x] += red[threadIdx.x + k];
        __syncthreads();
    }
    if (threadIdx.x == 0) gr[j] = red[0] / (float)n;
}

__global__ void gat_vec_mlp(const float* __restrict__ v, const float* __restrict__ W,
                            const float* __restrict__ b, float* __restrict__ out,
                            int Kd, int Nd, int doElu) {
    int j = blockIdx.x * blockDim.x + threadIdx.x;
    if (j >= Nd) return;
    float acc = b[j];
    for (int k = 0; k < Kd; ++k) acc += v[k] * W[(size_t)k * Nd + j];
    out[j] = doElu ? gat_elu(acc) : acc;
}

__global__ __launch_bounds__(1024)
void gat_softmax1d(const float* __restrict__ s, float* __restrict__ o, int n) {
    __shared__ float red[1024];
    int t = threadIdx.x;
    float v = (t < n) ? s[t] : -3.4e38f;
    red[t] = v; __syncthreads();
    for (int k = 512; k > 0; k >>= 1) {
        if (t < k) red[t] = fmaxf(red[t], red[t + k]);
        __syncthreads();
    }
    float mx = red[0]; __syncthreads();
    float e = (t < n) ? expf(v - mx) : 0.0f;
    red[t] = e; __syncthreads();
    for (int k = 512; k > 0; k >>= 1) {
        if (t < k) red[t] += red[t + k];
        __syncthreads();
    }
    if (t < n) o[t] = e / red[0];
}

// ================= host side =================
static inline int gat_ceil(long long a, long long b) { return (int)((a + b - 1) / b); }

extern "C" void kernel_launch(void* const* d_in, const int* in_sizes, int n_in,
                              void* d_out, int out_size, void* d_ws, size_t ws_size,
                              hipStream_t stream) {
    const int ND = 34, ED = 10, OUT = 1000;
    const int D = GAT_D, H = GAT_H, L = GAT_L;
    const int Nn = in_sizes[0] / ND;       // 20000
    const int E  = in_sizes[1] / ED;       // 160000

    const float* x         = (const float*)d_in[0];
    const float* edge_attr = (const float*)d_in[1];
    const int*   edge_idx  = (const int*)  d_in[2];
    const float* W_node = (const float*)d_in[3];
    const float* b_node = (const float*)d_in[4];
    const float* W_edge = (const float*)d_in[5];
    const float* b_edge = (const float*)d_in[6];
    const float* Wl  = (const float*)d_in[7];
    const float* bl  = (const float*)d_in[8];
    const float* Wr  = (const float*)d_in[9];
    const float* br  = (const float*)d_in[10];
    const float* We  = (const float*)d_in[11];
    const float* att = (const float*)d_in[12];
    const float* bg  = (const float*)d_in[13];
    const float* Wres = (const float*)d_in[14];
    const float* bres = (const float*)d_in[15];
    const float* Wh1 = (const float*)d_in[16];
    const float* bh1 = (const float*)d_in[17];
    const float* Wh2 = (const float*)d_in[18];
    const float* bh2 = (const float*)d_in[19];
    const float* Wh3 = (const float*)d_in[20];
    const float* bh3 = (const float*)d_in[21];
    float* out = (float*)d_out;

    // -------- carve workspace --------
    size_t off = 0;
    char* base = (char*)d_ws;
    auto carve = [&](size_t bytes) -> char* {
        char* p = base + off;
        off += (bytes + 255) & ~(size_t)255;
        return p;
    };
    const size_t WDD = (size_t)L * D * D;
    unsigned short* wl_bf   = (unsigned short*)carve(WDD * 2);   // transposed [n][k]
    unsigned short* wr_bf   = (unsigned short*)carve(WDD * 2);
    unsigned short* we_bf   = (unsigned short*)carve(WDD * 2);
    unsigned short* wres_bf = (unsigned short*)carve(WDD * 2);
    float*          hF   = (float*)         carve((size_t)Nn * D * 4);
    unsigned short* hBf  = (unsigned short*)carve((size_t)Nn * D * 2);
    unsigned short* eeBf = (unsigned short*)carve((size_t)E * D * 2);   // e_enc bf16
    float*          xl   = (float*)         carve((size_t)Nn * D * 4);
    float*          xr   = (float*)         carve((size_t)Nn * D * 4);
    float*          hres = (float*)         carve((size_t)Nn * D * 4);
    unsigned short* eel  = (unsigned short*)carve((size_t)E * D * 2);   // ee per layer
    float*          logi = (float*)         carve((size_t)E * H * 4);
    unsigned int*   lmax = (unsigned int*)  carve((size_t)Nn * H * 4);
    float*          deno = (float*)         carve((size_t)Nn * H * 4);
    float*          agg  = (float*)         carve((size_t)Nn * D * 4);
    float*          gr   = (float*)         carve((size_t)D * 4);
    float*          z1   = (float*)         carve((size_t)2 * D * 4);
    float*          z2   = (float*)         carve((size_t)D * 4);
    float*          sc   = (float*)         carve((size_t)OUT * 4);
    (void)ws_size; (void)n_in; (void)out_size;

    // -------- one-time weight conversion to bf16 (transposed) --------
    {
        int blk = 256;
        gat_cvt_bf16_T<<<gat_ceil(WDD, blk), blk, 0, stream>>>(Wl,   wl_bf,   WDD);
        gat_cvt_bf16_T<<<gat_ceil(WDD, blk), blk, 0, stream>>>(Wr,   wr_bf,   WDD);
        gat_cvt_bf16_T<<<gat_ceil(WDD, blk), blk, 0, stream>>>(We,   we_bf,   WDD);
        gat_cvt_bf16_T<<<gat_ceil(WDD, blk), blk, 0, stream>>>(Wres, wres_bf, WDD);
    }

    // -------- encoders --------
    {
        size_t tn = (size_t)Nn * D;
        gat_encoder<<<gat_ceil(tn, 256), 256, 0, stream>>>(
            x, W_node, b_node, hF, hBf, Nn, ND, D);
        size_t te = (size_t)E * D;
        gat_encoder<<<gat_ceil(te, 256), 256, 0, stream>>>(
            edge_attr, W_edge, b_edge, nullptr, eeBf, E, ED, D);
    }

    dim3 gemmBlk(128);
    dim3 gNodes(D / GT_BN, gat_ceil(Nn, GT_BM));
    dim3 gEdges(D / GT_BN, gat_ceil(E, GT_BM));

    // -------- GATv2 layers --------
    for (int i = 0; i < L; ++i) {
        const unsigned short* wlb = wl_bf   + (size_t)i * D * D;
        const unsigned short* wrb = wr_bf   + (size_t)i * D * D;
        const unsigned short* web = we_bf   + (size_t)i * D * D;
        const unsigned short* wsb = wres_bf + (size_t)i * D * D;

        gat_wmma_gemm<<<gNodes, gemmBlk, 0, stream>>>(
            hBf, wlb, bl + (size_t)i * D, xl, nullptr, Nn, D, D);
        gat_wmma_gemm<<<gNodes, gemmBlk, 0, stream>>>(
            hBf, wrb, br + (size_t)i * D, xr, nullptr, Nn, D, D);
        gat_wmma_gemm<<<gEdges, gemmBlk, 0, stream>>>(
            eeBf, web, nullptr, nullptr, eel, E, D, D);

        size_t nh = (size_t)Nn * H;
        gat_zero_f32<<<gat_ceil(nh, 256), 256, 0, stream>>>((float*)lmax, nh);
        gat_zero_f32<<<gat_ceil(nh, 256), 256, 0, stream>>>(deno, nh);
        size_t nd = (size_t)Nn * D;
        gat_zero_f32<<<gat_ceil(nd, 256), 256, 0, stream>>>(agg, nd);

        gat_edge_logits<<<E, 256, 0, stream>>>(
            edge_idx, xl, xr, eel, att + (size_t)i * H * GAT_C, logi, lmax, E);
        size_t eh = (size_t)E * H;
        gat_edge_softmax<<<gat_ceil(eh, 256), 256, 0, stream>>>(
            edge_idx, logi, lmax, deno, E);
        gat_edge_aggregate<<<E, 256, 0, stream>>>(
            edge_idx, logi, deno, xl, agg, E);

        gat_wmma_gemm<<<gNodes, gemmBlk, 0, stream>>>(
            hBf, wsb, bres + (size_t)i * D, hres, nullptr, Nn, D, D);
        gat_node_update<<<gat_ceil(nd, 256), 256, 0, stream>>>(
            agg, bg + (size_t)i * D, hres, hF, hBf, nd);
    }

    // -------- readout head --------
    gat_col_mean<<<D, 256, 0, stream>>>(hF, gr, Nn);
    gat_vec_mlp<<<gat_ceil(2 * D, 256), 256, 0, stream>>>(gr, Wh1, bh1, z1, D, 2 * D, 1);
    gat_vec_mlp<<<gat_ceil(D, 256),     256, 0, stream>>>(z1, Wh2, bh2, z2, 2 * D, D, 1);
    gat_vec_mlp<<<gat_ceil(OUT, 256),   256, 0, stream>>>(z2, Wh3, bh3, sc, D, OUT, 0);
    gat_softmax1d<<<1, 1024, 0, stream>>>(sc, out, OUT);
}
